// OuterProductMean_49804440764766
// MI455X (gfx1250) — matrix operand order repaired and checked by reading, not compiled, not run on hardware
//
#include <hip/hip_runtime.h>
#include <hip/hip_bf16.h>

typedef __attribute__((ext_vector_type(16))) _Float16 v16h;
typedef __attribute__((ext_vector_type(8)))  float    v8f;
typedef _Float16 h16;

#define S_DIM 128
#define N_DIM 256
#define CM    256
#define CH    32
#define CZ    128
#define LN_EPS_F 1e-5f
#define OPM_EPS_F 1e-3f

// workspace layout (bytes)
static constexpr size_t OFF_AT  = 0;         // h16 aT[256][32][128]   (2 MB)
static constexpr size_t OFF_BT  = 2097152;   // h16 bT[256][32][128]   (2 MB)
static constexpr size_t OFF_WT  = 4194304;   // h16 wT[2][32][256]     (32 KB)
static constexpr size_t OFF_BH  = 4227072;   // h16 biasH[64]          (128 B)
static constexpr size_t OFF_WOT = 4227200;   // h16 woT[128][1024]     (256 KB)

union FragU { v16h v; uint4 q[2]; };

// A-matrix fragment (f16 16x32): lane L: row = L%16; g = L/16;
// halves 0..7  -> K = k0 + g*8 + 0..7
// halves 8..15 -> K = k0 + g*8 + 16..23
__device__ inline v16h load_frag_a_g(const h16* __restrict__ base, int ld, int k0) {
  int lane = threadIdx.x & 31;
  const h16* p = base + (size_t)(lane & 15) * ld + k0 + ((lane >> 4) << 3);
  FragU f;
  f.q[0] = *(const uint4*)(p);
  f.q[1] = *(const uint4*)(p + 16);
  return f.v;
}

// B-matrix fragment (f16 32x16) from transposed storage baseT[n][k] (row stride ld):
// lane L: col n = n0 + L%16; halves 0..7 -> K = k0 + (L/16)*8 + 0..7, halves 8..15 -> K = k0 + (L/16)*8 + 16..23
__device__ inline v16h load_frag_bT(const h16* __restrict__ baseT, int ld, int k0, int n0) {
  int lane = threadIdx.x & 31;
  const h16* p = baseT + (size_t)(n0 + (lane & 15)) * ld + k0 + ((lane >> 4) << 3);
  FragU f;
  f.q[0] = *(const uint4*)(p);
  f.q[1] = *(const uint4*)(p + 16);
  return f.v;
}

__device__ inline v8f wmma_f16(v16h a, v16h b, v8f c) {
  return __builtin_amdgcn_wmma_f32_16x16x32_f16(false, a, false, b, (short)0, c, false, false);
}

// ---------------- Kernel 0: weight prep (transpose + f16 convert) ----------------
__global__ void opm_prep(const float* __restrict__ w1, const float* __restrict__ w2,
                         const float* __restrict__ b1, const float* __restrict__ b2,
                         const float* __restrict__ wo,
                         h16* __restrict__ wT, h16* __restrict__ biasH, h16* __restrict__ woT) {
  int idx = blockIdx.x * 256 + threadIdx.x;
  if (idx < 16384) {                    // wT[sel][n][k] = w[k][n]
    int sel = idx >> 13;
    int rem = idx & 8191;
    int n = rem >> 8;
    int k = rem & 255;
    const float* w = sel ? w2 : w1;
    wT[idx] = (h16)w[k * CH + n];
  } else if (idx < 16384 + 131072) {    // woT[z][cd] = wo[cd][z]
    int mi = idx - 16384;
    int z  = mi >> 10;
    int cd = mi & 1023;
    woT[mi] = (h16)wo[(size_t)cd * CZ + z];
  } else if (idx < 16384 + 131072 + 64) {
    int b = idx - 147456;
    int sel = b >> 5;
    int n = b & 31;
    biasH[b] = (h16)(sel ? b2[n] : b1[n]);
  }
}

// ---------------- Kernel 1: LayerNorm + dual projection (WMMA) ----------------
// Block = 16 rows (one s, 16 consecutive i). 128 threads = 4 waves.
__global__ __launch_bounds__(128) void opm_ln_proj(
    const float* __restrict__ m, const float* __restrict__ mask,
    const float* __restrict__ ln_w, const float* __restrict__ ln_b,
    const h16* __restrict__ wT, const h16* __restrict__ biasH,
    h16* __restrict__ aT, h16* __restrict__ bT) {
  __shared__ __align__(16) h16 lnT[16 * 256];
  __shared__ float maskL[16];
  int t   = threadIdx.x;
  int blk = blockIdx.x;
  int s   = blk >> 4;
  int i0  = (blk & 15) << 4;
  int r   = t >> 3;           // row 0..15
  int seg = (t & 7) << 5;     // 32-col segment

  const float* mrow = m + ((size_t)s * N_DIM + i0 + r) * CM + seg;
  float x[32];
  float sum = 0.f, sq = 0.f;
#pragma unroll
  for (int k = 0; k < 8; ++k) {
    float4 f = ((const float4*)mrow)[k];
    x[4*k+0] = f.x; x[4*k+1] = f.y; x[4*k+2] = f.z; x[4*k+3] = f.w;
    sum += f.x + f.y + f.z + f.w;
    sq  += f.x*f.x + f.y*f.y + f.z*f.z + f.w*f.w;
  }
#pragma unroll
  for (int off = 1; off < 8; off <<= 1) {  // 8 threads per row, same wave
    sum += __shfl_xor(sum, off);
    sq  += __shfl_xor(sq,  off);
  }
  float mean = sum * (1.0f / CM);
  float var  = sq * (1.0f / CM) - mean * mean;
  float rstd = rsqrtf(var + LN_EPS_F);
#pragma unroll
  for (int k = 0; k < 8; ++k) {
    float4 w  = ((const float4*)(ln_w + seg))[k];
    float4 bb = ((const float4*)(ln_b + seg))[k];
    lnT[r*256 + seg + 4*k+0] = (h16)((x[4*k+0] - mean) * rstd * w.x + bb.x);
    lnT[r*256 + seg + 4*k+1] = (h16)((x[4*k+1] - mean) * rstd * w.y + bb.y);
    lnT[r*256 + seg + 4*k+2] = (h16)((x[4*k+2] - mean) * rstd * w.z + bb.z);
    lnT[r*256 + seg + 4*k+3] = (h16)((x[4*k+3] - mean) * rstd * w.w + bb.w);
  }
  if ((t & 7) == 0) maskL[r] = mask[(size_t)s * N_DIM + i0 + r];
  __syncthreads();

  // [16,256] x [256,32] for each of w1/w2: wave -> (sel, n-tile)
  int wave = t >> 5;
  int lane = t & 31;
  int sel  = wave >> 1;
  int n0   = (wave & 1) << 4;
  const h16* wbase = wT + (size_t)sel * 32 * 256;
  v8f acc = {};
#pragma unroll
  for (int kk = 0; kk < 8; ++kk) {
    v16h af = load_frag_a_g(lnT, 256, kk * 32);
    v16h bf = load_frag_bT(wbase, 256, kk * 32, n0);
    acc = wmma_f16(af, bf, acc);
  }
  int n = n0 + (lane & 15);
  int g = lane >> 4;
  float bias = (float)biasH[sel * 32 + n];
  h16* dst = sel ? bT : aT;
#pragma unroll
  for (int v = 0; v < 8; ++v) {
    int rr = v + (g << 3);
    float val = (acc[v] + bias) * maskL[rr];
    // transposed layout: dst[i][c][s]
    dst[((size_t)(i0 + rr) * CH + n) * S_DIM + s] = (h16)val;
  }
}

// ---------------- Kernel 2: outer product (over s) + wo projection, fused ----------------
// Block = 16 i's x one j. 256 threads = 8 waves.
__global__ __launch_bounds__(256) void opm_outer(
    const h16* __restrict__ aT, const h16* __restrict__ bT,
    const h16* __restrict__ woT, const float* __restrict__ bo,
    const float* __restrict__ mask, float* __restrict__ out) {
  __shared__ __align__(16) h16 ldsC[16 * 32 * 32];  // C[i_local][c][d], 32 KB
  __shared__ float normL[16];
  int t    = threadIdx.x;
  int blk  = blockIdx.x;
  int j    = blk >> 4;
  int i0   = (blk & 15) << 4;
  int wave = t >> 5;
  int lane = t & 31;
  int g    = lane >> 4;

  if (t < 16) {  // norm[i,j] = sum_s mask[s,i]*mask[s,j] + eps
    float nv = OPM_EPS_F;
    const float* mi = mask + i0 + t;
    const float* mj = mask + j;
    for (int s = 0; s < S_DIM; ++s) nv = fmaf(mi[s * N_DIM], mj[s * N_DIM], nv);
    normL[t] = nv;
  }

  // Phase 1: C[(i,c), d] = sum_s aT[(i,c), s] * bT[j][d][s];  M=512, K=128, Nd=32
  const h16* aBase = aT + (size_t)i0 * CH * S_DIM;   // rows (i_local*32 + c), stride 128
  const h16* bBase = bT + (size_t)j * CH * S_DIM;    // rows d, stride 128
#pragma unroll
  for (int tile = 0; tile < 8; ++tile) {
    int tt = wave + (tile << 3);   // 64 tiles: mt(0..31) x dt(0..1)
    int mt = tt >> 1, dt = tt & 1;
    v8f acc = {};
#pragma unroll
    for (int kk = 0; kk < 4; ++kk) {
      v16h af = load_frag_a_g(aBase + (size_t)(mt << 4) * S_DIM, S_DIM, kk * 32);
      v16h bf = load_frag_bT(bBase, S_DIM, kk * 32, dt << 4);
      acc = wmma_f16(af, bf, acc);
    }
    int d = (dt << 4) + (lane & 15);
#pragma unroll
    for (int v = 0; v < 8; ++v) {
      int rrow = (mt << 4) + v + (g << 3);   // (i_local, c) flattened
      ldsC[rrow * 32 + d] = (h16)acc[v];
    }
  }
  __syncthreads();

  // Phase 2: out[i, j, z] = (C[i,:] . woT[z,:] + bo[z]) / norm;  [16,1024]x[1024,128]
  int z0 = wave << 4;
  v8f acc = {};
  for (int k0 = 0; k0 < 1024; k0 += 32) {
    v16h af = load_frag_a_g(ldsC, 1024, k0);
    v16h bf = load_frag_bT(woT, 1024, k0, z0);
    acc = wmma_f16(af, bf, acc);
  }
  int z = z0 + (lane & 15);
  float boz = bo[z];
#pragma unroll
  for (int v = 0; v < 8; ++v) {
    int rr = v + (g << 3);
    float val = (acc[v] + boz) / normL[rr];
    out[((size_t)(i0 + rr) * N_DIM + j) * CZ + z] = val;
  }
}

extern "C" void kernel_launch(void* const* d_in, const int* in_sizes, int n_in,
                              void* d_out, int out_size, void* d_ws, size_t ws_size,
                              hipStream_t stream) {
  (void)in_sizes; (void)n_in; (void)out_size; (void)ws_size;
  const float* m    = (const float*)d_in[0];
  const float* mask = (const float*)d_in[1];
  const float* ln_w = (const float*)d_in[2];
  const float* ln_b = (const float*)d_in[3];
  const float* w1   = (const float*)d_in[4];
  const float* b1   = (const float*)d_in[5];
  const float* w2   = (const float*)d_in[6];
  const float* b2   = (const float*)d_in[7];
  const float* wo   = (const float*)d_in[8];
  const float* bo   = (const float*)d_in[9];
  float* out = (float*)d_out;
  char* ws = (char*)d_ws;
  h16* aT  = (h16*)(ws + OFF_AT);
  h16* bT  = (h16*)(ws + OFF_BT);
  h16* wT  = (h16*)(ws + OFF_WT);
  h16* bH  = (h16*)(ws + OFF_BH);
  h16* woT = (h16*)(ws + OFF_WOT);

  opm_prep<<<577, 256, 0, stream>>>(w1, w2, b1, b2, wo, wT, bH, woT);
  opm_ln_proj<<<2048, 128, 0, stream>>>(m, mask, ln_w, ln_b, wT, bH, aT, bT);
  opm_outer<<<4096, 256, 0, stream>>>(aT, bT, woT, bo, mask, out);
}
